// Splat2D_45217415692676
// MI455X (gfx1250) — matrix operand order repaired and checked by reading, not compiled, not run on hardware
//
#include <hip/hip_runtime.h>

typedef _Float16 v8h  __attribute__((ext_vector_type(8)));
typedef _Float16 v16h __attribute__((ext_vector_type(16)));
typedef float    v8f  __attribute__((ext_vector_type(8)));

#define N_B    4
#define P_PTS  1024
#define C_CH   3
#define H_IMG  128
#define W_IMG  128
#define KSTEP  32
#define HT     16

// D[(c,h), w] = sum_p (v[p,c]*wy[p,h]) * wx[p,w]   per batch n
// Block = (n, 16-row h tile); 8 waves; wave = one 16-col w tile x 3 channels.
__global__ __launch_bounds__(256) void splat2d_wmma_kernel(
    const float* __restrict__ coords,   // [N,P,2] (x,y)
    const float* __restrict__ values,   // [N,P,C]
    const float* __restrict__ sigma,    // [N,1]
    float* __restrict__ out)            // [N,C,H,W]
{
    // wx^T staged K-contiguous per column -> B fragment = two 16B reads at kA and kA+16
    __shared__ __attribute__((aligned(64))) _Float16 lBx[W_IMG][KSTEP]; // 8 KB
    __shared__ __attribute__((aligned(64))) float    lWy[HT][KSTEP];    // 2 KB
    __shared__ __attribute__((aligned(64))) _Float16 lA[C_CH][HT][KSTEP]; // 3 KB

    const int tid  = threadIdx.x;
    const int wave = tid >> 5;
    const int lane = tid & 31;
    const int n    = blockIdx.x >> 3;          // /(H/HT)
    const int h0   = (blockIdx.x & 7) * HT;

    const float sg     = sigma[n];
    const float inv2s2 = 0.5f / (sg * sg);

    const int w0   = wave * 16;                // this wave's w tile
    const int mrow = lane & 15;                // M (A/D) or N (B) index
    const int kA   = (lane >> 4) * 8;          // A-fragment K base (0 or 8)
    const int kB   = (lane >> 4) * 16;         // 0 or 16; unused (B fragment also uses kA)

    v8f acc0 = {}, acc1 = {}, acc2 = {};

    for (int k0 = 0; k0 < P_PTS; k0 += KSTEP) {
        if ((k0 + KSTEP < P_PTS) && lane == 0) {
            __builtin_prefetch(&coords[((size_t)n * P_PTS + k0 + KSTEP) * 2], 0, 0);
            __builtin_prefetch(&values[((size_t)n * P_PTS + k0 + KSTEP) * C_CH], 0, 0);
        }

        // ---- phase 1: wx^T (B operand) and wy ----
        for (int idx = tid; idx < W_IMG * KSTEP; idx += 256) {
            const int w  = idx >> 5;
            const int kk = idx & 31;           // constant per thread -> x load CSE'd
            const float x  = coords[((size_t)n * P_PTS + k0 + kk) * 2 + 0];
            const float dx = (float)w - x;
            lBx[w][kk] = (_Float16)__expf(-dx * dx * inv2s2);
        }
        for (int idx = tid; idx < HT * KSTEP; idx += 256) {
            const int m  = idx >> 5;
            const int kk = idx & 31;
            const float y  = coords[((size_t)n * P_PTS + k0 + kk) * 2 + 1];
            const float dy = (float)(h0 + m) - y;
            lWy[m][kk] = __expf(-dy * dy * inv2s2);
        }
        __syncthreads();

        // ---- phase 2: A[c][m][kk] = v[p][c] * wy[m][kk] ----
        for (int idx = tid; idx < C_CH * HT * KSTEP; idx += 256) {
            const int c  = idx >> 9;
            const int r  = idx & 511;
            const int m  = r >> 5;
            const int kk = r & 31;
            const float v = values[((size_t)n * P_PTS + k0 + kk) * C_CH + c];
            lA[c][m][kk] = (_Float16)(v * lWy[m][kk]);
        }
        __syncthreads();

        // ---- phase 3: WMMA (ISA fragment layouts, wave32) ----
        const v8h blo = *(const v8h*)&lBx[w0 + mrow][kA];
        const v8h bhi = *(const v8h*)&lBx[w0 + mrow][kA + 16];
        const v16h b = __builtin_shufflevector(blo, bhi,
                     0,1,2,3,4,5,6,7,8,9,10,11,12,13,14,15);

        v8h a0lo = *(const v8h*)&lA[0][mrow][kA];
        v8h a0hi = *(const v8h*)&lA[0][mrow][kA + 16];
        v16h a0  = __builtin_shufflevector(a0lo, a0hi,
                     0,1,2,3,4,5,6,7,8,9,10,11,12,13,14,15);
        acc0 = __builtin_amdgcn_wmma_f32_16x16x32_f16(
                   false, a0, false, b, (short)0, acc0, false, false);

        v8h a1lo = *(const v8h*)&lA[1][mrow][kA];
        v8h a1hi = *(const v8h*)&lA[1][mrow][kA + 16];
        v16h a1  = __builtin_shufflevector(a1lo, a1hi,
                     0,1,2,3,4,5,6,7,8,9,10,11,12,13,14,15);
        acc1 = __builtin_amdgcn_wmma_f32_16x16x32_f16(
                   false, a1, false, b, (short)0, acc1, false, false);

        v8h a2lo = *(const v8h*)&lA[2][mrow][kA];
        v8h a2hi = *(const v8h*)&lA[2][mrow][kA + 16];
        v16h a2  = __builtin_shufflevector(a2lo, a2hi,
                     0,1,2,3,4,5,6,7,8,9,10,11,12,13,14,15);
        acc2 = __builtin_amdgcn_wmma_f32_16x16x32_f16(
                   false, a2, false, b, (short)0, acc2, false, false);

        __syncthreads();   // protect LDS before next iteration's overwrite
    }

    // ---- epilogue: C/D layout -> VGPR r holds (M = r + 8*(lane>=16), N = lane%16)
    const int colg  = w0 + mrow;
    const int rbase = h0 + ((lane >> 4) * 8);
    float* o0 = out + (((size_t)n * C_CH + 0) * H_IMG) * W_IMG;
    float* o1 = out + (((size_t)n * C_CH + 1) * H_IMG) * W_IMG;
    float* o2 = out + (((size_t)n * C_CH + 2) * H_IMG) * W_IMG;
#pragma unroll
    for (int r = 0; r < 8; ++r) {
        const size_t off = (size_t)(rbase + r) * W_IMG + colg;
        o0[off] = acc0[r];
        o1[off] = acc1[r];
        o2[off] = acc2[r];
    }
}

extern "C" void kernel_launch(void* const* d_in, const int* in_sizes, int n_in,
                              void* d_out, int out_size, void* d_ws, size_t ws_size,
                              hipStream_t stream) {
    (void)in_sizes; (void)n_in; (void)out_size; (void)d_ws; (void)ws_size;
    const float* coords = (const float*)d_in[0];
    const float* values = (const float*)d_in[1];
    const float* sigma  = (const float*)d_in[2];
    float* out = (float*)d_out;

    dim3 grid(N_B * (H_IMG / HT));   // 32 blocks
    dim3 block(256);                 // 8 wave32s
    splat2d_wmma_kernel<<<grid, block, 0, stream>>>(coords, values, sigma, out);
}
